// RNNLayer_61203283968022
// MI455X (gfx1250) — matrix operand order repaired and checked by reading, not compiled, not run on hardware
//
#include <hip/hip_runtime.h>
#include <hip/hip_bf16.h>
#include <stdint.h>

// ---------------------------------------------------------------------------
// Persistent 2-layer GRU + projection for MI455X (gfx1250, wave32, WMMA).
// bf16 GEMMs (v_wmma_f32_16x16x32_bf16) with fp32 accumulation + fp32 state.
// 32 persistent workgroups x 384 threads (12 waves); weights staged in LDS.
// ---------------------------------------------------------------------------

typedef __attribute__((ext_vector_type(16))) __bf16 v16bf;
typedef __attribute__((ext_vector_type(8)))  float  v8f;

#define B_   64
#define T_   1024
#define IN_  256
#define H_   512
#define OUT_ 256
#define NWG  32
#define NTHR 384   // 12 waves of 32

union FragCast {
    struct { uint4 lo, hi; } u;   // 32 bytes
    v16bf v;
};

// A fragment: 16x32 bf16, A[m][k]. lane L: m=m0+(L&15); slots 0..7 = k0+(L>>4)*8 .. +7,
// slots 8..15 = k0+16+(L>>4)*8 .. +7  (ISA 7.12.2, 16-bit A 16x32 layout).
__device__ __forceinline__ v16bf ld_frag_a(const __hip_bfloat16* __restrict__ base,
                                           size_t ld, int m0, int k0, int lane) {
    int m = m0 + (lane & 15);
    int k = k0 + ((lane >> 4) << 3);
    const __hip_bfloat16* p = base + (size_t)m * ld + k;
    FragCast f;
    f.u.lo = *reinterpret_cast<const uint4*>(p);
    f.u.hi = *reinterpret_cast<const uint4*>(p + 16);
    return f.v;
}

// B fragment: 32x16 bf16, Bmat[k][n] = W[n][k] (row-major weight slice, 16 rows x K).
// lane L: n=(L&15); slots 0..7 = k0+(L>>4)*8 .. +7, slots 8..15 = k0+16+(L>>4)*8 .. +7.
__device__ __forceinline__ v16bf ld_frag_b(const __hip_bfloat16* __restrict__ wslice,
                                           int K, int k0, int lane) {
    int n = lane & 15;
    int k = k0 + ((lane >> 4) << 3);
    const __hip_bfloat16* p = wslice + (size_t)n * K + k;
    FragCast f;
    f.u.lo = *reinterpret_cast<const uint4*>(p);
    f.u.hi = *reinterpret_cast<const uint4*>(p + 16);
    return f.v;
}

__device__ __forceinline__ v8f wmma_bf16(v16bf a, v16bf b, v8f c) {
    return __builtin_amdgcn_wmma_f32_16x16x32_bf16(false, a, false, b, (short)0, c,
                                                   false, false);
}

__device__ __forceinline__ float sigm(float x) { return 1.0f / (1.0f + __expf(-x)); }

// Grid-wide sense barrier: monotonically increasing device-scope counter.
__device__ __forceinline__ void grid_barrier(unsigned* cnt, unsigned* round) {
    __threadfence();          // make this thread's global writes visible (agent scope)
    __syncthreads();
    if (threadIdx.x == 0) {
        unsigned r = ++(*round);
        __hip_atomic_fetch_add(cnt, 1u, __ATOMIC_ACQ_REL, __HIP_MEMORY_SCOPE_AGENT);
        unsigned target = r * (unsigned)NWG;
        while (__hip_atomic_load(cnt, __ATOMIC_ACQUIRE, __HIP_MEMORY_SCOPE_AGENT) < target)
            __builtin_amdgcn_s_sleep(1);
    }
    __syncthreads();
}

// Stage this WG's 48-row (3 gates x 16 cols) bf16 weight slice into LDS.
__device__ void stage_gate_w(__hip_bfloat16* dst, const __hip_bfloat16* src,
                             int K, int wg_col0) {
    int vpr = K >> 3;                 // uint4 (8 bf16) per row
    int total = 48 * vpr;
    for (int i = threadIdx.x; i < total; i += NTHR) {
        int r = i / vpr, v = i - r * vpr;
        int gt = r >> 4, c = r & 15;
        reinterpret_cast<uint4*>(dst)[i] =
            reinterpret_cast<const uint4*>(src)[((size_t)gt * H_ + wg_col0 + c) * vpr + v];
    }
}

__device__ void stage_proj_w(__hip_bfloat16* dst, const __hip_bfloat16* src, int wg_col0) {
    int vpr = H_ >> 3;
    int total = 16 * vpr;
    for (int i = threadIdx.x; i < total; i += NTHR) {
        int r = i / vpr, v = i - r * vpr;
        reinterpret_cast<uint4*>(dst)[i] =
            reinterpret_cast<const uint4*>(src)[((size_t)wg_col0 + r) * vpr + v];
    }
}

// One GRU layer phase for this WG's 16 hidden columns:
// gi = A_in @ Wih^T + b_ih (input part), gh = A_h @ Whh^T + b_hh (hidden part),
// then r/z/n gate math and the h update into the opposite parity buffers.
__device__ void gate_gemm_update(
    const __hip_bfloat16* __restrict__ Ain, size_t ldA, int Kin,
    const __hip_bfloat16* __restrict__ Ah,  int Kh,
    const __hip_bfloat16* w_in, const __hip_bfloat16* w_h,     // LDS slices [48][K]
    const float* __restrict__ b_ih, const float* __restrict__ b_hh,
    int wg_col0,
    float* gi, float* gh,                                      // LDS [3][64][16]
    const float* __restrict__ hprev,
    float* __restrict__ hnewf, __hip_bfloat16* __restrict__ hnewb) {

    const int tid  = threadIdx.x;
    const int wave = tid >> 5;
    const int lane = tid & 31;

    if (wave < 12) {
        const int gt = wave >> 2;          // gate: 0=r 1=z 2=n
        const int m0 = (wave & 3) << 4;    // batch-row tile
        const __hip_bfloat16* wi = w_in + (size_t)gt * 16 * Kin;
        const __hip_bfloat16* wh = w_h  + (size_t)gt * 16 * Kh;

        v8f ci0 = {}, ci1 = {};
        for (int kb = 0; kb < Kin; kb += 64) {
            ci0 = wmma_bf16(ld_frag_a(Ain, ldA, m0, kb,      lane),
                            ld_frag_b(wi, Kin, kb,      lane), ci0);
            ci1 = wmma_bf16(ld_frag_a(Ain, ldA, m0, kb + 32, lane),
                            ld_frag_b(wi, Kin, kb + 32, lane), ci1);
        }
        v8f ch0 = {}, ch1 = {};
        for (int kb = 0; kb < Kh; kb += 64) {
            ch0 = wmma_bf16(ld_frag_a(Ah, (size_t)Kh, m0, kb,      lane),
                            ld_frag_b(wh, Kh, kb,      lane), ch0);
            ch1 = wmma_bf16(ld_frag_a(Ah, (size_t)Kh, m0, kb + 32, lane),
                            ld_frag_b(wh, Kh, kb + 32, lane), ch1);
        }
        v8f ci = ci0 + ci1;
        v8f ch = ch0 + ch1;

        int n    = lane & 15;
        int half = lane >> 4;
        float bi = b_ih[gt * H_ + wg_col0 + n];
        float bh = b_hh[gt * H_ + wg_col0 + n];
        // C/D layout: element (m,n) -> vgpr v, m = m0 + half*8 + v, lane n = lane&15
        float* gip = gi + ((size_t)gt * 64 + m0 + half * 8) * 16 + n;
        float* ghp = gh + ((size_t)gt * 64 + m0 + half * 8) * 16 + n;
#pragma unroll
        for (int v = 0; v < 8; ++v) {
            gip[v * 16] = ci[v] + bi;
            ghp[v * 16] = ch[v] + bh;
        }
    }
    __syncthreads();

    // Elementwise GRU update over this WG's [64 x 16] tile.
    for (int idx = tid; idx < 64 * 16; idx += NTHR) {
        int m = idx >> 4, n = idx & 15;
        float r  = sigm(gi[idx]        + gh[idx]);
        float z  = sigm(gi[1024 + idx] + gh[1024 + idx]);
        float nn = tanhf(gi[2048 + idx] + r * gh[2048 + idx]);
        int col  = wg_col0 + n;
        float hp = hprev[(size_t)m * H_ + col];
        float hv = (1.0f - z) * nn + z * hp;
        hnewf[(size_t)m * H_ + col] = hv;
        hnewb[(size_t)m * H_ + col] = __float2bfloat16(hv);
    }
    __syncthreads();   // protect gi/gh before next phase overwrites
}

__global__ void __launch_bounds__(NTHR, 1) gru_persistent(
    const __hip_bfloat16* __restrict__ xb,
    const __hip_bfloat16* __restrict__ wih0, const __hip_bfloat16* __restrict__ whh0,
    const __hip_bfloat16* __restrict__ wih1, const __hip_bfloat16* __restrict__ whh1,
    const __hip_bfloat16* __restrict__ wproj,
    const float* __restrict__ b_ih0, const float* __restrict__ b_hh0,
    const float* __restrict__ b_ih1, const float* __restrict__ b_hh1,
    const float* __restrict__ b_proj,
    float* __restrict__ h0f, float* __restrict__ h1f,          // [2][64][512]
    __hip_bfloat16* __restrict__ h0b, __hip_bfloat16* __restrict__ h1b,
    float* __restrict__ y, unsigned* __restrict__ cnt) {

    extern __shared__ unsigned char smem_raw[];
    __hip_bfloat16* wA = reinterpret_cast<__hip_bfloat16*>(smem_raw); // [48][256]
    __hip_bfloat16* wB = wA + 48 * IN_;                               // [48][512]
    __hip_bfloat16* wC = wB + 48 * H_;                                // [48][512]
    __hip_bfloat16* wD = wC + 48 * H_;                                // [48][512]
    __hip_bfloat16* wP = wD + 48 * H_;                                // [16][512]
    float* gi = reinterpret_cast<float*>(wP + 16 * H_);               // [3][64][16]
    float* gh = gi + 3 * 64 * 16;

    const int wg      = blockIdx.x;
    const int wg_col0 = wg << 4;

    stage_gate_w(wA, wih0, IN_, wg_col0);
    stage_gate_w(wB, whh0, H_,  wg_col0);
    stage_gate_w(wC, wih1, H_,  wg_col0);
    stage_gate_w(wD, whh1, H_,  wg_col0);
    if (wg < 16) stage_proj_w(wP, wproj, wg_col0);
    __syncthreads();

    unsigned round = 0;
    const size_t HB = (size_t)B_ * H_;

    for (int t = 0; t < T_; ++t) {
        const int p = t & 1;

        // ---- layer 0: x_t (K=256) + h0 (K=512) -> h0' ----
        gate_gemm_update(xb + (size_t)t * IN_, (size_t)T_ * IN_, IN_,
                         h0b + (size_t)p * HB, H_,
                         wA, wB, b_ih0, b_hh0, wg_col0, gi, gh,
                         h0f + (size_t)p * HB,
                         h0f + (size_t)(p ^ 1) * HB, h0b + (size_t)(p ^ 1) * HB);
        grid_barrier(cnt, &round);

        // ---- layer 1: h0' (K=512) + h1 (K=512) -> h1' ----
        gate_gemm_update(h0b + (size_t)(p ^ 1) * HB, H_, H_,
                         h1b + (size_t)p * HB, H_,
                         wC, wD, b_ih1, b_hh1, wg_col0, gi, gh,
                         h1f + (size_t)p * HB,
                         h1f + (size_t)(p ^ 1) * HB, h1b + (size_t)(p ^ 1) * HB);
        grid_barrier(cnt, &round);

        // ---- projection: y_t = h1' @ Wproj^T + b_proj (WGs 0..15, waves 0..3) ----
        if (wg < 16) {
            const int wave = threadIdx.x >> 5;
            const int lane = threadIdx.x & 31;
            if (wave < 4) {
                const int m0 = wave << 4;
                const __hip_bfloat16* h1n = h1b + (size_t)(p ^ 1) * HB;
                v8f c0 = {}, c1 = {};
                for (int kb = 0; kb < H_; kb += 64) {
                    c0 = wmma_bf16(ld_frag_a(h1n, H_, m0, kb,      lane),
                                   ld_frag_b(wP, H_, kb,      lane), c0);
                    c1 = wmma_bf16(ld_frag_a(h1n, H_, m0, kb + 32, lane),
                                   ld_frag_b(wP, H_, kb + 32, lane), c1);
                }
                v8f c = c0 + c1;
                int n = lane & 15, half = lane >> 4;
                float bp = b_proj[wg_col0 + n];
#pragma unroll
                for (int v = 0; v < 8; ++v) {
                    int m = m0 + half * 8 + v;
                    y[((size_t)m * T_ + t) * OUT_ + wg_col0 + n] = c[v] + bp;
                }
            }
        }
        // no barrier needed: next phase A touches only h0 buffers / y untouched
    }
}

// ---------------------------------------------------------------------------
__global__ void cvt_bf16(const float* __restrict__ s, __hip_bfloat16* __restrict__ d,
                         int n) {
    int i = blockIdx.x * blockDim.x + threadIdx.x;
    int stride = gridDim.x * blockDim.x;
    for (; i < n; i += stride) d[i] = __float2bfloat16(s[i]);
}

__global__ void init_state(const float* __restrict__ h00, const float* __restrict__ h01,
                           float* h0f, float* h1f,
                           __hip_bfloat16* h0b, __hip_bfloat16* h1b, unsigned* cnt) {
    int i = blockIdx.x * blockDim.x + threadIdx.x;
    if (i == 0) *cnt = 0u;
    if (i < B_ * H_) {
        int col = i & (H_ - 1);
        float a = h00[col], b = h01[col];
        h0f[i] = a; h0b[i] = __float2bfloat16(a);
        h1f[i] = b; h1b[i] = __float2bfloat16(b);
    }
}

extern "C" void kernel_launch(void* const* d_in, const int* in_sizes, int n_in,
                              void* d_out, int out_size, void* d_ws, size_t ws_size,
                              hipStream_t stream) {
    const float* x      = (const float*)d_in[0];
    const float* W_ih0  = (const float*)d_in[1];
    const float* W_hh0  = (const float*)d_in[2];
    const float* b_ih0  = (const float*)d_in[3];
    const float* b_hh0  = (const float*)d_in[4];
    const float* W_ih1  = (const float*)d_in[5];
    const float* W_hh1  = (const float*)d_in[6];
    const float* b_ih1  = (const float*)d_in[7];
    const float* b_hh1  = (const float*)d_in[8];
    const float* W_proj = (const float*)d_in[9];
    const float* b_proj = (const float*)d_in[10];
    const float* h0_0   = (const float*)d_in[11];
    const float* h0_1   = (const float*)d_in[12];
    float* y = (float*)d_out;

    // Workspace carve-out (256 B aligned regions).
    uint8_t* ws = (uint8_t*)d_ws;
    size_t off = 0;
    auto carve = [&](size_t bytes) -> uint8_t* {
        off = (off + 255) & ~(size_t)255;
        uint8_t* p = ws + off;
        off += bytes;
        return p;
    };
    unsigned*        cnt   = (unsigned*)carve(sizeof(unsigned));
    __hip_bfloat16*  xb    = (__hip_bfloat16*)carve((size_t)B_ * T_ * IN_ * 2);
    __hip_bfloat16*  wih0b = (__hip_bfloat16*)carve((size_t)3 * H_ * IN_ * 2);
    __hip_bfloat16*  whh0b = (__hip_bfloat16*)carve((size_t)3 * H_ * H_ * 2);
    __hip_bfloat16*  wih1b = (__hip_bfloat16*)carve((size_t)3 * H_ * H_ * 2);
    __hip_bfloat16*  whh1b = (__hip_bfloat16*)carve((size_t)3 * H_ * H_ * 2);
    __hip_bfloat16*  wpb   = (__hip_bfloat16*)carve((size_t)OUT_ * H_ * 2);
    float*           h0f   = (float*)carve((size_t)2 * B_ * H_ * 4);
    float*           h1f   = (float*)carve((size_t)2 * B_ * H_ * 4);
    __hip_bfloat16*  h0b   = (__hip_bfloat16*)carve((size_t)2 * B_ * H_ * 2);
    __hip_bfloat16*  h1b   = (__hip_bfloat16*)carve((size_t)2 * B_ * H_ * 2);

    // bf16 conversions (cheap vs. the recurrent loop; L2-resident afterwards)
    cvt_bf16<<<4096, 256, 0, stream>>>(x,      xb,    B_ * T_ * IN_);
    cvt_bf16<<<512,  256, 0, stream>>>(W_ih0,  wih0b, 3 * H_ * IN_);
    cvt_bf16<<<512,  256, 0, stream>>>(W_hh0,  whh0b, 3 * H_ * H_);
    cvt_bf16<<<512,  256, 0, stream>>>(W_ih1,  wih1b, 3 * H_ * H_);
    cvt_bf16<<<512,  256, 0, stream>>>(W_hh1,  whh1b, 3 * H_ * H_);
    cvt_bf16<<<256,  256, 0, stream>>>(W_proj, wpb,   OUT_ * H_);
    init_state<<<(B_ * H_ + 255) / 256, 256, 0, stream>>>(h0_0, h0_1, h0f, h1f,
                                                          h0b, h1b, cnt);

    // LDS: 4 gate-weight slices + proj slice + gi/gh scratch = 212,992 B (< 320 KB/WGP)
    size_t smem = (size_t)(48 * IN_ + 3 * 48 * H_ + 16 * H_) * 2 + 2 * 3 * 64 * 16 * 4;
    gru_persistent<<<NWG, NTHR, smem, stream>>>(
        xb, wih0b, whh0b, wih1b, whh1b, wpb,
        b_ih0, b_hh0, b_ih1, b_hh1, b_proj,
        h0f, h1f, h0b, h1b, y, cnt);
}
